// CausalAttention_69879117906636
// MI455X (gfx1250) — compile-verified, matrix operand order checked by reading, not run on hardware
//
#include <hip/hip_runtime.h>

// ---------------------------------------------------------------------------
// Causal attention for MI455X (gfx1250, wave32, WMMA + Tensor Data Mover).
// Pipeline (bf16 storage as u16 bit patterns, fp32 accumulate):
//   1) Q/K/V = x @ W{q,k,v}        (fp32 -> bf16 packed-pair fill, WMMA GEMM)
//   2) S     = scale * Q @ K^T     (tiles via TDM tensor_load_to_lds,
//                                   causal upper tiles skipped)
//   3) P     = causal softmax(S)   (in place, zeros above diagonal)
//   4) out   = P @ V               (P tile via TDM, K-loop causally truncated)
// Compute-bound (~1 TFLOP vs ~13us of HBM traffic at 23.3 TB/s), so every
// matmul runs through v_wmma_f32_16x16x32_bf16.
// ---------------------------------------------------------------------------

typedef unsigned short u16;
typedef __attribute__((ext_vector_type(16))) __bf16 v16bf;
typedef __attribute__((ext_vector_type(8)))  float  v8f;
typedef __attribute__((ext_vector_type(4)))  unsigned int u32x4;
typedef __attribute__((ext_vector_type(8)))  int i32x8;
typedef __attribute__((ext_vector_type(4)))  int i32x4;

#define TB   256   // threads per block (8 wave32)
#define BM   128   // block tile M
#define BN   128   // block tile N
#define BK    32   // block tile K (= WMMA K for bf16)
#define LDST  40   // padded LDS row stride (elements): 64B row + 16B pad

#if defined(__has_builtin)
#if __has_builtin(__builtin_amdgcn_tensor_load_to_lds) && \
    __has_builtin(__builtin_amdgcn_s_wait_tensorcnt)
#define USE_TDM 1
#endif
#endif

__device__ __forceinline__ u16 f2bf(float f) {
  unsigned u = __float_as_uint(f);
  unsigned r = 0x7FFFu + ((u >> 16) & 1u);   // round-to-nearest-even
  return (u16)((u + r) >> 16);
}
__device__ __forceinline__ float bf2f(u16 h) {
  return __uint_as_float(((unsigned)h) << 16);
}
__device__ __forceinline__ unsigned pack_bf2(float lo, float hi) {
  return (unsigned)f2bf(lo) | ((unsigned)f2bf(hi) << 16);
}

// Build a 16x32 bf16 WMMA operand fragment from an LDS tile (row-major,
// stride LDST). Per ISA 16-bit A-layout: lanes 0-15 -> rows 0-15 with
// K = {0..7,16..23}; lanes 16-31 -> same rows with K += 8. Pairs of
// consecutive K are contiguous, so read them as 32-bit words.
__device__ __forceinline__ v16bf lds_frag16x32(const u16* tile, int lane) {
  const int row  = lane & 15;
  const int koff = (lane & 16) ? 8 : 0;
  union { v16bf v; unsigned int u[8]; } f;
  const u16* rp = tile + row * LDST + koff;
#pragma unroll
  for (int p = 0; p < 8; ++p) {
    const int k = ((p & 3) << 1) + ((p >> 2) << 4);   // 0,2,4,6,16,18,20,22
    f.u[p] = *reinterpret_cast<const unsigned int*>(rp + k);
  }
  return f.v;
}

#ifdef USE_TDM
// Issue one TDM descriptor: load a 128-row x 32-col bf16 tile (rows
// contiguous in global, pitch row_stride_elems) into LDS at lds_byte,
// padded 16B per 64B row -> LDS row stride of 40 elements (matches LDST).
// D# bitfields per CDNA5 ISA. This toolchain
// exposes the 6-arg builtin: (u32x4, i32x8, i32x4, i32x4, i32x8, cpol).
__device__ __forceinline__ void tdm_load_tile128x32_bf16(
    const u16* g, unsigned lds_byte, unsigned row_stride_elems) {
  unsigned long long ga = (unsigned long long)(size_t)g;
  u32x4 g0 = { 0x1u,                                     // count=1, user D#
               lds_byte,                                 // lds_addr (bytes)
               (unsigned)ga,                             // global_addr[31:0]
               ((unsigned)(ga >> 32) & 0x01FFFFFFu)      // global_addr[56:32]
                   | (2u << 30) };                       // type = 2 (image)
  const int w0 = (1 << 16)      // data_size = 1 -> 2 bytes
               | (1 << 20)      // pad_enable
               | (3 << 22)      // pad_interval = 3 -> every 16 DWORDs (64B)
               | (3 << 25);     // pad_amount   = 3 -> 4 DWORDs (16B)
  i32x8 g1 = { w0,
               (int)(32u << 16),            // tensor_dim0[15:0] = 32
               (int)(128u << 16),           // tensor_dim1[15:0] = 128
               (int)(32u << 16),            // tile_dim0 = 32
               128,                         // tile_dim1 = 128
               (int)row_stride_elems,       // tensor_dim0_stride[31:0]
               0, 0 };
  i32x4 gz4 = { 0, 0, 0, 0 };               // 2D tensor: groups 2/3 unused
  i32x8 gz8 = { 0, 0, 0, 0, 0, 0, 0, 0 };
  __builtin_amdgcn_tensor_load_to_lds(g0, g1, gz4, gz4, gz8, 0);
}
#endif

// ---------------------------------------------------------------------------
// Kernel 1: Out[M,N](bf16) = X[M,K](f32) @ W[K,N](f32)
// ---------------------------------------------------------------------------
__global__ __launch_bounds__(TB)
void qkv_gemm_f32_bf16(const float* __restrict__ X, const float* __restrict__ W,
                       u16* __restrict__ Out, int M, int Kd, int N) {
  __shared__ u16 At[BM * LDST];   // A tile, row-major [m][k]
  __shared__ u16 Bt[BN * LDST];   // B tile, N-major   [n][k]
  const int t    = threadIdx.x;
  const int lane = t & 31;
  const int wave = t >> 5;
  const int wm   = wave >> 2;                 // 0..1 -> 64-row strip
  const int wn   = wave & 3;                  // 0..3 -> 32-col strip
  const int tileM = blockIdx.y * BM;
  const int tileN = blockIdx.x * BN;

  v8f zero = {};
  v8f acc[4][2];
#pragma unroll
  for (int f = 0; f < 4; ++f)
#pragma unroll
    for (int g = 0; g < 2; ++g) acc[f][g] = zero;

  const int ksteps = Kd / BK;
  for (int kb = 0; kb < ksteps; ++kb) {
#pragma unroll
    for (int i = 0; i < (BM * BK) / (2 * TB); ++i) {  // A: f32 pairs -> packed bf16
      int idx = i * TB + t;
      int r = idx >> 4;                               // 16 pairs per row
      int c = (idx & 15) << 1;
      const float* xp = &X[(size_t)(tileM + r) * Kd + kb * BK + c];
      *reinterpret_cast<unsigned*>(&At[r * LDST + c]) = pack_bf2(xp[0], xp[1]);
    }
#pragma unroll
    for (int i = 0; i < (BK * BN) / (2 * TB); ++i) {  // Bt[n][k] = W[k][n], k-pairs
      int idx = i * TB + t;
      int k2 = idx >> 7;                              // pair index 0..15
      int n  = idx & 127;
      const float* wp = &W[(size_t)(kb * BK + 2 * k2) * N + tileN + n];
      *reinterpret_cast<unsigned*>(&Bt[n * LDST + 2 * k2]) =
          pack_bf2(wp[0], wp[(size_t)N]);
    }
    __syncthreads();
    if (kb + 1 < ksteps) {                            // global_prefetch_b8 next slab
      __builtin_prefetch(&X[(size_t)(tileM + (t & 127)) * Kd + (kb + 1) * BK], 0, 0);
      if (t < BK)
        __builtin_prefetch(&W[(size_t)((kb + 1) * BK + t) * N + tileN], 0, 0);
    }
    v16bf a[4], b[2];
#pragma unroll
    for (int f = 0; f < 4; ++f)
      a[f] = lds_frag16x32(&At[(wm * 64 + f * 16) * LDST], lane);
#pragma unroll
    for (int g = 0; g < 2; ++g)
      b[g] = lds_frag16x32(&Bt[(wn * 32 + g * 16) * LDST], lane);
#pragma unroll
    for (int f = 0; f < 4; ++f)
#pragma unroll
      for (int g = 0; g < 2; ++g)
        acc[f][g] = __builtin_amdgcn_wmma_f32_16x16x32_bf16(
            false, a[f], false, b[g], (short)0, acc[f][g], false, false);
    __syncthreads();
  }

  const int rbase = (lane & 16) ? 8 : 0;              // C/D layout: VGPR r -> M=r(+8)
  const int col0  = lane & 15;
#pragma unroll
  for (int f = 0; f < 4; ++f)
#pragma unroll
    for (int g = 0; g < 2; ++g)
#pragma unroll
      for (int r = 0; r < 8; ++r) {
        int row = tileM + wm * 64 + f * 16 + rbase + r;
        int col = tileN + wn * 32 + g * 16 + col0;
        Out[(size_t)row * N + col] = f2bf(acc[f][g][r]);
      }
}

// ---------------------------------------------------------------------------
// Kernel 2: Sc[b,i,j](bf16) = scale * Q[b,i,:] . K[b,j,:]
// Both tiles are contiguous bf16 row blocks -> loaded by the TDM.
// Tiles entirely above the causal diagonal are skipped.
// ---------------------------------------------------------------------------
__global__ __launch_bounds__(TB)
void scores_gemm(const u16* __restrict__ Q, const u16* __restrict__ K,
                 u16* __restrict__ Sc, int S, int D, float scale) {
  const int tileM = blockIdx.y * BM;                  // query rows i
  const int tileN = blockIdx.x * BN;                  // key cols  j
  if (tileN > tileM) return;                          // fully masked tile

  __shared__ u16 At[BM * LDST];
  __shared__ u16 Bt[BN * LDST];
  const int t    = threadIdx.x;
  const int lane = t & 31;
  const int wave = t >> 5;
  const int wm   = wave >> 2;
  const int wn   = wave & 3;
  const int b    = blockIdx.z;
  const u16* Qb = Q + (size_t)b * S * D;
  const u16* Kb = K + (size_t)b * S * D;
#ifdef USE_TDM
  const unsigned ldsA = (unsigned)(size_t)(const void*)At;  // addr[31:0] = LDS byte
  const unsigned ldsB = (unsigned)(size_t)(const void*)Bt;
#endif

  v8f zero = {};
  v8f acc[4][2];
#pragma unroll
  for (int f = 0; f < 4; ++f)
#pragma unroll
    for (int g = 0; g < 2; ++g) acc[f][g] = zero;

  const int ksteps = D / BK;
  for (int kb = 0; kb < ksteps; ++kb) {
#ifdef USE_TDM
    if (wave == 0) {                                  // one DMA issue per tile
      tdm_load_tile128x32_bf16(&Qb[(size_t)tileM * D + kb * BK], ldsA, (unsigned)D);
      tdm_load_tile128x32_bf16(&Kb[(size_t)tileN * D + kb * BK], ldsB, (unsigned)D);
      __builtin_amdgcn_s_wait_tensorcnt(0);
    }
#else
#pragma unroll
    for (int i = 0; i < (BM * BK) / TB; ++i) {        // A = Q rows
      int idx = i * TB + t;
      int r = idx >> 5, c = idx & 31;
      At[r * LDST + c] = Qb[(size_t)(tileM + r) * D + kb * BK + c];
    }
#pragma unroll
    for (int i = 0; i < (BN * BK) / TB; ++i) {        // Bt[n=j][k=d] = K[j][d]
      int idx = i * TB + t;
      int n = idx >> 5, c = idx & 31;
      Bt[n * LDST + c] = Kb[(size_t)(tileN + n) * D + kb * BK + c];
    }
#endif
    __syncthreads();
    v16bf a[4], bfr[2];
#pragma unroll
    for (int f = 0; f < 4; ++f)
      a[f] = lds_frag16x32(&At[(wm * 64 + f * 16) * LDST], lane);
#pragma unroll
    for (int g = 0; g < 2; ++g)
      bfr[g] = lds_frag16x32(&Bt[(wn * 32 + g * 16) * LDST], lane);
#pragma unroll
    for (int f = 0; f < 4; ++f)
#pragma unroll
      for (int g = 0; g < 2; ++g)
        acc[f][g] = __builtin_amdgcn_wmma_f32_16x16x32_bf16(
            false, a[f], false, bfr[g], (short)0, acc[f][g], false, false);
    __syncthreads();
  }

  const int rbase = (lane & 16) ? 8 : 0;
  const int col0  = lane & 15;
  u16* Sb = Sc + (size_t)b * S * S;
#pragma unroll
  for (int f = 0; f < 4; ++f)
#pragma unroll
    for (int g = 0; g < 2; ++g)
#pragma unroll
      for (int r = 0; r < 8; ++r) {
        int row = tileM + wm * 64 + f * 16 + rbase + r;
        int col = tileN + wn * 32 + g * 16 + col0;
        Sb[(size_t)row * S + col] = f2bf(acc[f][g][r] * scale);
      }
}

// ---------------------------------------------------------------------------
// Kernel 3: in-place causal softmax over each score row; zeros for j > i so
// the P@V GEMM can run dense up to the causal limit.
// ---------------------------------------------------------------------------
__global__ __launch_bounds__(TB)
void softmax_causal(u16* __restrict__ Sc, int S) {
  __shared__ float red[TB];
  const int b = blockIdx.y;
  const int i = blockIdx.x;
  const int t = threadIdx.x;
  u16* row = Sc + ((size_t)b * S + i) * S;
  const int n = i + 1;                                // valid keys: j <= i

  float m = -3.0e38f;
  for (int j = t; j < n; j += TB) m = fmaxf(m, bf2f(row[j]));
  red[t] = m; __syncthreads();
  for (int s = TB / 2; s > 0; s >>= 1) {
    if (t < s) red[t] = fmaxf(red[t], red[t + s]);
    __syncthreads();
  }
  m = red[0]; __syncthreads();

  float sum = 0.0f;
  for (int j = t; j < n; j += TB) sum += __expf(bf2f(row[j]) - m);
  red[t] = sum; __syncthreads();
  for (int s = TB / 2; s > 0; s >>= 1) {
    if (t < s) red[t] += red[t + s];
    __syncthreads();
  }
  const float inv = 1.0f / red[0];

  for (int j = t; j < n; j += TB) row[j] = f2bf(__expf(bf2f(row[j]) - m) * inv);
  for (int j = n + t; j < S; j += TB) row[j] = (u16)0;  // zero masked region
}

// ---------------------------------------------------------------------------
// Kernel 4: out[b,i,:](f32) = P[b,i,:] @ V[b,:,:]; K-loop stops at causal
// limit of the query tile (P is zero beyond it). P tile is a contiguous row
// block -> TDM; V tile is a transpose gather -> packed manual fill.
// ---------------------------------------------------------------------------
__global__ __launch_bounds__(TB)
void attn_out_gemm(const u16* __restrict__ P, const u16* __restrict__ V,
                   float* __restrict__ O, int S, int D) {
  __shared__ u16 At[BM * LDST];
  __shared__ u16 Bt[BN * LDST];
  const int t    = threadIdx.x;
  const int lane = t & 31;
  const int wave = t >> 5;
  const int wm   = wave >> 2;
  const int wn   = wave & 3;
  const int tileM = blockIdx.y * BM;
  const int tileN = blockIdx.x * BN;
  const int b     = blockIdx.z;
  const u16* Pb = P + (size_t)b * S * S;
  const u16* Vb = V + (size_t)b * S * D;
#ifdef USE_TDM
  const unsigned ldsA = (unsigned)(size_t)(const void*)At;
#endif

  v8f zero = {};
  v8f acc[4][2];
#pragma unroll
  for (int f = 0; f < 4; ++f)
#pragma unroll
    for (int g = 0; g < 2; ++g) acc[f][g] = zero;

  const int ksteps = (tileM + BM) / BK;               // keys <= tileM + 127 only
  for (int kb = 0; kb < ksteps; ++kb) {
#ifdef USE_TDM
    if (wave == 0) {
      tdm_load_tile128x32_bf16(&Pb[(size_t)tileM * S + kb * BK], ldsA, (unsigned)S);
      __builtin_amdgcn_s_wait_tensorcnt(0);
    }
#else
#pragma unroll
    for (int i = 0; i < (BM * BK) / TB; ++i) {        // A = P rows
      int idx = i * TB + t;
      int r = idx >> 5, c = idx & 31;
      At[r * LDST + c] = Pb[(size_t)(tileM + r) * S + kb * BK + c];
    }
#endif
#pragma unroll
    for (int i = 0; i < (BK * BN) / (2 * TB); ++i) {  // Bt[n=d][k=key], k-pairs
      int idx = i * TB + t;
      int k2 = idx >> 7;                              // pair index 0..15
      int n  = idx & 127;
      const u16* vp = &Vb[(size_t)(kb * BK + 2 * k2) * D + tileN + n];
      *reinterpret_cast<unsigned*>(&Bt[n * LDST + 2 * k2]) =
          (unsigned)vp[0] | ((unsigned)vp[(size_t)D] << 16);
    }
    __syncthreads();
    if (kb + 1 < ksteps && t < BK)                    // prefetch next V slab
      __builtin_prefetch(&Vb[(size_t)((kb + 1) * BK + t) * D + tileN], 0, 0);
    v16bf a[4], bfr[2];
#pragma unroll
    for (int f = 0; f < 4; ++f)
      a[f] = lds_frag16x32(&At[(wm * 64 + f * 16) * LDST], lane);
#pragma unroll
    for (int g = 0; g < 2; ++g)
      bfr[g] = lds_frag16x32(&Bt[(wn * 32 + g * 16) * LDST], lane);
#pragma unroll
    for (int f = 0; f < 4; ++f)
#pragma unroll
      for (int g = 0; g < 2; ++g)
        acc[f][g] = __builtin_amdgcn_wmma_f32_16x16x32_bf16(
            false, a[f], false, bfr[g], (short)0, acc[f][g], false, false);
    __syncthreads();
  }

  const int rbase = (lane & 16) ? 8 : 0;
  const int col0  = lane & 15;
  float* Ob = O + (size_t)b * S * D;
#pragma unroll
  for (int f = 0; f < 4; ++f)
#pragma unroll
    for (int g = 0; g < 2; ++g)
#pragma unroll
      for (int r = 0; r < 8; ++r) {
        int row = tileM + wm * 64 + f * 16 + rbase + r;
        int col = tileN + wn * 32 + g * 16 + col0;
        Ob[(size_t)row * D + col] = acc[f][g][r];
      }
}

// ---------------------------------------------------------------------------
extern "C" void kernel_launch(void* const* d_in, const int* in_sizes, int n_in,
                              void* d_out, int out_size, void* d_ws, size_t ws_size,
                              hipStream_t stream) {
  (void)in_sizes; (void)n_in; (void)out_size; (void)ws_size;
  constexpr int Bb = 4, Ss = 4096, Dd = 2048;
  constexpr float scale = 0.022097086912079608f;      // 1/sqrt(2048)

  const float* x  = (const float*)d_in[0];
  const float* Wq = (const float*)d_in[1];
  const float* Wk = (const float*)d_in[2];
  const float* Wv = (const float*)d_in[3];
  float* out = (float*)d_out;

  // Workspace: Q,K,V bf16 (3 * 64 MB) + scores/P bf16 in place (128 MB)
  const size_t MD = (size_t)Bb * Ss * Dd;
  u16* Qb = (u16*)d_ws;
  u16* Kb = Qb + MD;
  u16* Vb = Kb + MD;
  u16* Sc = Vb + MD;

  dim3 blk(TB);
  dim3 g1(Dd / BN, (Bb * Ss) / BM);
  qkv_gemm_f32_bf16<<<g1, blk, 0, stream>>>(x, Wq, Qb, Bb * Ss, Dd, Dd);
  qkv_gemm_f32_bf16<<<g1, blk, 0, stream>>>(x, Wk, Kb, Bb * Ss, Dd, Dd);
  qkv_gemm_f32_bf16<<<g1, blk, 0, stream>>>(x, Wv, Vb, Bb * Ss, Dd, Dd);

  dim3 g2(Ss / BN, Ss / BM, Bb);
  scores_gemm<<<g2, blk, 0, stream>>>(Qb, Kb, Sc, Ss, Dd, scale);

  dim3 g3(Ss, Bb);
  softmax_causal<<<g3, blk, 0, stream>>>(Sc, Ss);

  dim3 g4(Dd / BN, Ss / BM, Bb);
  attn_out_gemm<<<g4, blk, 0, stream>>>(Sc, Vb, out, Ss, Dd);
}
